// CusparseDynamicLinear_73718818668813
// MI455X (gfx1250) — matrix operand order repaired and checked by reading, not compiled, not run on hardware
//
#include <hip/hip_runtime.h>

// ---------------------------------------------------------------------------
// out = data @ (mask*weight)^T + bias   (fp32 in/out)
// fp32 emulated via bf16 hi/lo split: a*w ~= ah*wh + ah*wl + al*wh
// using V_WMMA_F32_16X16X32_BF16 (CDNA5, wave32).
//
// Fast path (needs ~192MiB workspace):
//   prep kernels split data / mask*weight into bf16 hi+lo planes once,
//   then a pure-bf16 GEMM with ping-pong LDS fed by
//   GLOBAL_LOAD_ASYNC_TO_LDS_B128 (ASYNCcnt-tracked, no VGPR staging).
// Fallback: fused conversion GEMM.
// ---------------------------------------------------------------------------

typedef __bf16 bf16;
typedef __attribute__((ext_vector_type(4)))  bf16  v4bf;
typedef __attribute__((ext_vector_type(8)))  bf16  v8bf;
typedef __attribute__((ext_vector_type(16))) bf16  v16bf;
typedef __attribute__((ext_vector_type(4)))  float v4f;
typedef __attribute__((ext_vector_type(8)))  float v8f;

#define BM 128          // C-tile rows per block
#define BN 128          // C-tile cols per block
#define BK 32           // K chunk (one bf16 WMMA depth)
#define SR 40           // padded LDS row stride (bf16): 80B rows -> conflict-free b128

__device__ __forceinline__ v16bf cat8(v8bf a, v8bf b) {
    return __builtin_shufflevector(a, b, 0,1,2,3,4,5,6,7,8,9,10,11,12,13,14,15);
}

// global (GVS: sgpr base + vgpr u32 byte-offset) -> LDS, 16B, async (ASYNCcnt)
__device__ __forceinline__ void async_b128(unsigned lds_addr, unsigned gvoff,
                                           const void* base) {
    asm volatile("global_load_async_to_lds_b128 %0, %1, %2"
                 :: "v"(lds_addr), "v"(gvoff), "s"(base) : "memory");
}
// same, +16B on both LDS and memory address via instruction offset
__device__ __forceinline__ void async_b128_o16(unsigned lds_addr, unsigned gvoff,
                                               const void* base) {
    asm volatile("global_load_async_to_lds_b128 %0, %1, %2 offset:16"
                 :: "v"(lds_addr), "v"(gvoff), "s"(base) : "memory");
}
__device__ __forceinline__ void wait_async0() {
    asm volatile("s_wait_asynccnt 0x0" ::: "memory");
}

// ============================ prep kernels =================================

__global__ __launch_bounds__(256, 1)
void prep_split_A(const float* __restrict__ A, bf16* __restrict__ hi,
                  bf16* __restrict__ lo) {
    const size_t i = ((size_t)blockIdx.x * 256 + threadIdx.x) * 8;
    v4f x0 = *(const v4f*)(A + i);
    v4f x1 = *(const v4f*)(A + i + 4);
    v8bf h, l;
#pragma unroll
    for (int j = 0; j < 4; ++j) {
        const float a0 = x0[j], a1 = x1[j];
        const bf16 h0 = (bf16)a0, h1 = (bf16)a1;
        h[j]     = h0; l[j]     = (bf16)(a0 - (float)h0);
        h[j + 4] = h1; l[j + 4] = (bf16)(a1 - (float)h1);
    }
    *(v8bf*)(hi + i) = h;
    *(v8bf*)(lo + i) = l;
}

__global__ __launch_bounds__(256, 1)
void prep_split_W(const float* __restrict__ W, const float* __restrict__ M,
                  bf16* __restrict__ hi, bf16* __restrict__ lo) {
    const size_t i = ((size_t)blockIdx.x * 256 + threadIdx.x) * 8;
    v4f w0 = *(const v4f*)(W + i);
    v4f w1 = *(const v4f*)(W + i + 4);
    v4f m0 = *(const v4f*)(M + i);
    v4f m1 = *(const v4f*)(M + i + 4);
    v8bf h, l;
#pragma unroll
    for (int j = 0; j < 4; ++j) {
        const float a0 = w0[j] * m0[j], a1 = w1[j] * m1[j];
        const bf16 h0 = (bf16)a0, h1 = (bf16)a1;
        h[j]     = h0; l[j]     = (bf16)(a0 - (float)h0);
        h[j + 4] = h1; l[j + 4] = (bf16)(a1 - (float)h1);
    }
    *(v8bf*)(hi + i) = h;
    *(v8bf*)(lo + i) = l;
}

// ========== fast GEMM: pure bf16, ping-pong LDS, async global->LDS =========

__global__ __launch_bounds__(256, 1)
void gemm_wmma_bf16x3(const bf16* __restrict__ Ahi, const bf16* __restrict__ Alo,
                      const bf16* __restrict__ Whi, const bf16* __restrict__ Wlo,
                      const float* __restrict__ bias, float* __restrict__ C,
                      int K, int N) {
    // 2 ping-pong buffers x 4 planes x 128 x SR bf16 = 80KB LDS
    __shared__ bf16 lds[2][4][BM * SR];   // planes: 0=Ahi 1=Alo 2=Whi 3=Wlo

    const int tid   = threadIdx.x;
    const int lane  = tid & 31;
    const int wave  = tid >> 5;
    const int waveM = wave >> 2;     // 0..1
    const int waveN = wave & 3;      // 0..3

    const int blockN0 = blockIdx.x * BN;
    const int blockM0 = blockIdx.y * BM;

    v8f acc[4][2];
#pragma unroll
    for (int mt = 0; mt < 4; ++mt)
#pragma unroll
        for (int nt = 0; nt < 2; ++nt) { v8f z = {}; acc[mt][nt] = z; }

    // staging map: each thread owns 16 contiguous bf16 of one row per plane
    const int ldrow = tid >> 1;          // 0..127
    const int ldcol = (tid & 1) * 16;    // 0 or 16
    // per-thread byte offsets into each plane (fit in u32: planes <= 67MB)
    const unsigned aByte = (unsigned)(((size_t)(blockM0 + ldrow) * K + ldcol) * 2);
    const unsigned wByte = (unsigned)(((size_t)(blockN0 + ldrow) * K + ldcol) * 2);
    const void* gbase[4] = { Ahi, Alo, Whi, Wlo };
    const unsigned gvoff0[4] = { aByte, aByte, wByte, wByte };
    // per-thread LDS byte addresses of the staging slot, per plane, buffer 0
    unsigned ldsSlot[4];
#pragma unroll
    for (int p = 0; p < 4; ++p)
        ldsSlot[p] = (unsigned)(unsigned long long)&lds[0][p][ldrow * SR + ldcol];
    const unsigned bufStride = (unsigned)(4u * BM * SR * sizeof(bf16)); // 40960B

    const int half  = lane >> 4;
    const int l16   = lane & 15;
    const int akoff = half * 8;          // A frag K 0..7/16..23 vs 8..15/24..31
    const int bkoff = half * 8;          // B frag K 0..7/16..23 vs 8..15/24..31

    const int KC = K / BK;               // chunks of 32 (64B per row-chunk)

    // async-preload chunk 0 into buffer 0
#pragma unroll
    for (int p = 0; p < 4; ++p) {
        async_b128(ldsSlot[p], gvoff0[p], gbase[p]);
        async_b128_o16(ldsSlot[p], gvoff0[p], gbase[p]);
    }
    wait_async0();
    __syncthreads();

    for (int kc = 0; kc < KC; ++kc) {
        const int cur = kc & 1;

        // prefetch distance 2 into L2
        if (kc + 2 < KC) {
            const unsigned kpf = (unsigned)((kc + 2) * BK * 2);
            __builtin_prefetch((const char*)gbase[0] + gvoff0[0] + kpf, 0, 1);
            __builtin_prefetch((const char*)gbase[1] + gvoff0[1] + kpf, 0, 1);
            __builtin_prefetch((const char*)gbase[2] + gvoff0[2] + kpf, 0, 1);
            __builtin_prefetch((const char*)gbase[3] + gvoff0[3] + kpf, 0, 1);
        }
        // async loads for next chunk straight into the other LDS buffer
        if (kc + 1 < KC) {
            const unsigned kb = (unsigned)((kc + 1) * BK * 2);   // byte advance in K
            const unsigned bofs = (unsigned)((cur ^ 1)) * bufStride;
#pragma unroll
            for (int p = 0; p < 4; ++p) {
                async_b128(ldsSlot[p] + bofs, gvoff0[p] + kb, gbase[p]);
                async_b128_o16(ldsSlot[p] + bofs, gvoff0[p] + kb, gbase[p]);
            }
        }

        // ---- fragments from current buffer ----
        v16bf a_hi[4], a_lo[4];
#pragma unroll
        for (int mt = 0; mt < 4; ++mt) {
            const int r = (waveM * 64 + mt * 16 + l16) * SR;
            a_hi[mt] = cat8(*(const v8bf*)(&lds[cur][0][r + akoff]),
                            *(const v8bf*)(&lds[cur][0][r + akoff + 16]));
            a_lo[mt] = cat8(*(const v8bf*)(&lds[cur][1][r + akoff]),
                            *(const v8bf*)(&lds[cur][1][r + akoff + 16]));
        }
        v16bf b_hi[2], b_lo[2];
#pragma unroll
        for (int nt = 0; nt < 2; ++nt) {
            const int r = (waveN * 32 + nt * 16 + l16) * SR;
            b_hi[nt] = cat8(*(const v8bf*)(&lds[cur][2][r + bkoff]),
                            *(const v8bf*)(&lds[cur][2][r + bkoff + 16]));
            b_lo[nt] = cat8(*(const v8bf*)(&lds[cur][3][r + bkoff]),
                            *(const v8bf*)(&lds[cur][3][r + bkoff + 16]));
        }

        // ---- 3-term bf16 emulation of fp32 MMA ----
#pragma unroll
        for (int mt = 0; mt < 4; ++mt)
#pragma unroll
            for (int nt = 0; nt < 2; ++nt) {
                acc[mt][nt] = __builtin_amdgcn_wmma_f32_16x16x32_bf16(
                    false, a_hi[mt], false, b_hi[nt], (short)0, acc[mt][nt], false, false);
                acc[mt][nt] = __builtin_amdgcn_wmma_f32_16x16x32_bf16(
                    false, a_hi[mt], false, b_lo[nt], (short)0, acc[mt][nt], false, false);
                acc[mt][nt] = __builtin_amdgcn_wmma_f32_16x16x32_bf16(
                    false, a_lo[mt], false, b_hi[nt], (short)0, acc[mt][nt], false, false);
            }

        // next chunk's asyncs must have landed before anyone reads that buffer
        wait_async0();
        __syncthreads();
    }

    // ---- bias-fused epilogue ----
#pragma unroll
    for (int nt = 0; nt < 2; ++nt) {
        const int gcol = blockN0 + waveN * 32 + nt * 16 + l16;
        const float bv = bias[gcol];
#pragma unroll
        for (int mt = 0; mt < 4; ++mt) {
            const int grow0 = blockM0 + waveM * 64 + mt * 16 + half * 8;
#pragma unroll
            for (int i = 0; i < 8; ++i)
                C[(size_t)(grow0 + i) * N + gcol] = acc[mt][nt][i] + bv;
        }
    }
}

// ================ fallback: fused-conversion GEMM ==========================

__global__ __launch_bounds__(256, 1)
void masked_linear_wmma_fused(const float* __restrict__ A, const float* __restrict__ W,
                              const float* __restrict__ Msk, const float* __restrict__ bias,
                              float* __restrict__ C, int K, int N) {
    __shared__ bf16 As_hi[BM * SR];
    __shared__ bf16 As_lo[BM * SR];
    __shared__ bf16 Ws_hi[BN * SR];
    __shared__ bf16 Ws_lo[BN * SR];

    const int tid = threadIdx.x, lane = tid & 31, wave = tid >> 5;
    const int waveM = wave >> 2, waveN = wave & 3;
    const int blockN0 = blockIdx.x * BN, blockM0 = blockIdx.y * BM;

    v8f acc[4][2];
#pragma unroll
    for (int mt = 0; mt < 4; ++mt)
#pragma unroll
        for (int nt = 0; nt < 2; ++nt) { v8f z = {}; acc[mt][nt] = z; }

    const int ldrow = tid >> 1, ldcol = (tid & 1) * 16;
    const float* Arow = A   + (size_t)(blockM0 + ldrow) * K + ldcol;
    const float* Wrow = W   + (size_t)(blockN0 + ldrow) * K + ldcol;
    const float* Mrow = Msk + (size_t)(blockN0 + ldrow) * K + ldcol;

    const int half = lane >> 4, l16 = lane & 15;
    const int akoff = half * 8, bkoff = half * 8;

    for (int k0 = 0; k0 < K; k0 += BK) {
#pragma unroll
        for (int i = 0; i < 4; ++i) {
            const int c = ldcol + i * 4;
            v4f av = *(const v4f*)(Arow + k0 + i * 4);
            v4f wv = *(const v4f*)(Wrow + k0 + i * 4);
            v4f mv = *(const v4f*)(Mrow + k0 + i * 4);
            v4bf ah, al, wh, wl;
#pragma unroll
            for (int j = 0; j < 4; ++j) {
                const float a = av[j];     const bf16 h1 = (bf16)a;
                ah[j] = h1; al[j] = (bf16)(a - (float)h1);
                const float w = wv[j] * mv[j]; const bf16 h2 = (bf16)w;
                wh[j] = h2; wl[j] = (bf16)(w - (float)h2);
            }
            *(v4bf*)(&As_hi[ldrow * SR + c]) = ah;
            *(v4bf*)(&As_lo[ldrow * SR + c]) = al;
            *(v4bf*)(&Ws_hi[ldrow * SR + c]) = wh;
            *(v4bf*)(&Ws_lo[ldrow * SR + c]) = wl;
        }
        __syncthreads();

        v16bf a_hi[4], a_lo[4];
#pragma unroll
        for (int mt = 0; mt < 4; ++mt) {
            const int r = (waveM * 64 + mt * 16 + l16) * SR;
            a_hi[mt] = cat8(*(const v8bf*)(&As_hi[r + akoff]),
                            *(const v8bf*)(&As_hi[r + akoff + 16]));
            a_lo[mt] = cat8(*(const v8bf*)(&As_lo[r + akoff]),
                            *(const v8bf*)(&As_lo[r + akoff + 16]));
        }
        v16bf b_hi[2], b_lo[2];
#pragma unroll
        for (int nt = 0; nt < 2; ++nt) {
            const int r = (waveN * 32 + nt * 16 + l16) * SR;
            b_hi[nt] = cat8(*(const v8bf*)(&Ws_hi[r + bkoff]),
                            *(const v8bf*)(&Ws_hi[r + bkoff + 16]));
            b_lo[nt] = cat8(*(const v8bf*)(&Ws_lo[r + bkoff]),
                            *(const v8bf*)(&Ws_lo[r + bkoff + 16]));
        }
#pragma unroll
        for (int mt = 0; mt < 4; ++mt)
#pragma unroll
            for (int nt = 0; nt < 2; ++nt) {
                acc[mt][nt] = __builtin_amdgcn_wmma_f32_16x16x32_bf16(
                    false, a_hi[mt], false, b_hi[nt], (short)0, acc[mt][nt], false, false);
                acc[mt][nt] = __builtin_amdgcn_wmma_f32_16x16x32_bf16(
                    false, a_hi[mt], false, b_lo[nt], (short)0, acc[mt][nt], false, false);
                acc[mt][nt] = __builtin_amdgcn_wmma_f32_16x16x32_bf16(
                    false, a_lo[mt], false, b_hi[nt], (short)0, acc[mt][nt], false, false);
            }
        __syncthreads();
    }

#pragma unroll
    for (int nt = 0; nt < 2; ++nt) {
        const int gcol = blockN0 + waveN * 32 + nt * 16 + l16;
        const float bv = bias[gcol];
#pragma unroll
        for (int mt = 0; mt < 4; ++mt) {
            const int grow0 = blockM0 + waveM * 64 + mt * 16 + half * 8;
#pragma unroll
            for (int i = 0; i < 8; ++i)
                C[(size_t)(grow0 + i) * N + gcol] = acc[mt][nt][i] + bv;
        }
    }
}

// ================================ launch ===================================

extern "C" void kernel_launch(void* const* d_in, const int* in_sizes, int n_in,
                              void* d_out, int out_size, void* d_ws, size_t ws_size,
                              hipStream_t stream) {
    const float* data   = (const float*)d_in[0];
    const float* weight = (const float*)d_in[1];
    const float* mask   = (const float*)d_in[2];
    const float* bias   = (const float*)d_in[3];
    float* out          = (float*)d_out;

    const int out_f  = in_sizes[3];              // 4096
    const int in_f   = in_sizes[1] / out_f;      // 4096
    const int tokens = in_sizes[0] / in_f;       // 8192

    const size_t aElems = (size_t)tokens * in_f;   // 33.5M
    const size_t wElems = (size_t)out_f * in_f;    // 16.8M
    const size_t need   = (2 * aElems + 2 * wElems) * sizeof(bf16);  // ~192 MiB

    dim3 grid(out_f / BN, tokens / BM);
    dim3 block(256);

    if (ws_size >= need) {
        bf16* Ahi = (bf16*)d_ws;
        bf16* Alo = Ahi + aElems;
        bf16* Whi = Alo + aElems;
        bf16* Wlo = Whi + wElems;

        prep_split_A<<<dim3((unsigned)(aElems / (256 * 8))), block, 0, stream>>>(
            data, Ahi, Alo);
        prep_split_W<<<dim3((unsigned)(wElems / (256 * 8))), block, 0, stream>>>(
            weight, mask, Whi, Wlo);
        gemm_wmma_bf16x3<<<grid, block, 0, stream>>>(
            Ahi, Alo, Whi, Wlo, bias, out, in_f, out_f);
    } else {
        masked_linear_wmma_fused<<<grid, block, 0, stream>>>(
            data, weight, mask, bias, out, in_f, out_f);
    }
}
